// SlidingWindowAttention_56684978372931
// MI455X (gfx1250) — matrix operand order repaired and checked by reading, not compiled, not run on hardware
//
#include <hip/hip_runtime.h>

// ---------------- problem constants ----------------
constexpr int S_LEN   = 4096;
constexpr int D_MODEL = 1024;
constexpr int N_HEADS = 16;
constexpr int HEAD_DIM = 64;
constexpr int HALF_W  = 256;          // WINDOW/2
constexpr float ATT_SCALE = 0.125f;   // 1/sqrt(64)
constexpr float NEG_BIG = -1.0e30f;

typedef __attribute__((ext_vector_type(16))) __bf16 v16bf;
typedef __attribute__((ext_vector_type(8)))  float  v8f;

struct alignas(16) U16x8 { unsigned short h[8]; };
union FragA {                       // A-matrix fragment: two 16B runs per lane
  v16bf v;
  U16x8 p[2];
  unsigned short h[16];
};

__device__ __forceinline__ v8f wmma_bf16(v16bf a, v16bf b, v8f c) {
#if defined(__gfx1250__)
  // D = A(16x32 bf16) * B(32x16 bf16) + C(16x16 f32)
  return __builtin_amdgcn_wmma_f32_16x16x32_bf16(false, a, false, b,
                                                 (short)0, c, false, false);
#else
  (void)a; (void)b;
  return c;
#endif
}

// Async global->LDS copy of 16 bytes per lane (ASYNCcnt-tracked TDM-lite path).
// Generic LDS pointer low 32 bits == LDS byte offset (shared aperture).
__device__ __forceinline__ void async_cp16(const unsigned short* g,
                                           unsigned short* l) {
#if defined(__gfx1250__)
  asm volatile("global_load_async_to_lds_b128 %0, %1, off"
               :: "v"((unsigned)(size_t)l),
                  "v"((unsigned long long)(size_t)g)
               : "memory");
#else
  for (int i = 0; i < 8; ++i) l[i] = g[i];
#endif
}

__device__ __forceinline__ void wait_async(int pending) {
#if defined(__gfx1250__)
  if (pending > 0)
    asm volatile("s_wait_asynccnt 8" ::: "memory");   // keep next tile in flight
  else
    asm volatile("s_wait_asynccnt 0" ::: "memory");
#else
  (void)pending;
#endif
}

__device__ __forceinline__ unsigned short f2bf(float f) {
  unsigned u = __float_as_uint(f);
  u += 0x7FFFu + ((u >> 16) & 1u);    // round-to-nearest-even
  return (unsigned short)(u >> 16);
}

__device__ __forceinline__ int imin(int a, int b) { return a < b ? a : b; }
__device__ __forceinline__ int imax(int a, int b) { return a > b ? a : b; }

// ---------------- stage 0: casts ----------------
__global__ void cast_f32_bf16(const float* __restrict__ src,
                              unsigned short* __restrict__ dst, int n) {
  int i = blockIdx.x * blockDim.x + threadIdx.x;
  if (i < n) dst[i] = f2bf(src[i]);
}

// w: [E][D] row-major (torch Linear weight) -> wt: [D][E] (k-major for WMMA B)
__global__ void cast_transpose_bf16(const float* __restrict__ w,
                                    unsigned short* __restrict__ wt,
                                    float scale) {
  int i = blockIdx.x * blockDim.x + threadIdx.x;   // over D*E = 1M
  int d = i >> 10;
  int e = i & 1023;
  wt[i] = f2bf(w[e * D_MODEL + d] * scale);
}

// ---------------- stage 1/3: bf16 GEMM  C = A[M,K] * Bt[K,N] ----------------
// 128x128 block tile (4 waves, 64x64 per wave), K-step 32.
// A/B tiles staged in LDS via async global->LDS copies, double buffered.
// MODE 0: bf16 out[row*N+col]   MODE 1: bf16 out[col*M+row] (transposed store)
// MODE 2: f32  out[row*N+col]
template <int MODE>
__global__ __launch_bounds__(128)
void gemm_bf16(const unsigned short* __restrict__ A,
               const unsigned short* __restrict__ Bt,
               void* __restrict__ outp, int M, int K, int N) {
  constexpr int ARS = 40;    // A LDS row stride (32 data + pad), elems
  constexpr int BRS = 136;   // B LDS row stride (128 data + pad), elems
  __shared__ alignas(32) unsigned short As[2][128 * ARS];
  __shared__ alignas(32) unsigned short Bs[2][32 * BRS];

  const int tid  = threadIdx.x;
  const int lane = tid & 31;
  const int wave = tid >> 5;
  const int h2   = lane >> 4;          // 0/1 half of wave
  const int l15  = lane & 15;
  const int wm   = wave & 1;
  const int wn   = wave >> 1;
  const int row0 = blockIdx.x * 128;
  const int col0 = blockIdx.y * 128;

  // stage one K-chunk (A: 128x32, B: 32x128) into LDS buffer `buf`
  // 8 async-copy instructions per wave per stage.
  auto stage = [&](int buf, int kc) {
#pragma unroll
    for (int p = 0; p < 4; ++p) {            // A tile: 512 x 16B chunks
      int id = tid + p * 128;
      int r = id >> 2, c = id & 3;
      async_cp16(A + (size_t)(row0 + r) * K + kc + c * 8,
                 &As[buf][r * ARS + c * 8]);
    }
#pragma unroll
    for (int p = 0; p < 4; ++p) {            // B tile: 512 x 16B chunks
      int id = tid + p * 128;
      int r = id >> 4, c = id & 15;
      async_cp16(Bt + (size_t)(kc + r) * N + col0 + c * 8,
                 &Bs[buf][r * BRS + c * 8]);
    }
  };

  v8f acc[4][4];
#pragma unroll
  for (int i = 0; i < 4; ++i)
#pragma unroll
    for (int j = 0; j < 4; ++j) acc[i][j] = {};

  const int nk = K / 32;
  stage(0, 0);
  for (int ki = 0; ki < nk; ++ki) {
    const int buf = ki & 1;
    if (ki + 1 < nk) {
      stage(buf ^ 1, (ki + 1) * 32);   // prefetch next tile (async)
      wait_async(8);                   // current tile's 8 oldest have landed
    } else {
      wait_async(0);
    }
    __syncthreads();

    FragA a[4], b[4];
#pragma unroll
    for (int i = 0; i < 4; ++i) {
      const unsigned short* ap = &As[buf][(wm * 64 + i * 16 + l15) * ARS + h2 * 8];
      a[i].p[0] = *(const U16x8*)(ap);        // K = h2*8 + 0..7
      a[i].p[1] = *(const U16x8*)(ap + 16);   // K = 16 + h2*8 + 0..7
    }
#pragma unroll
    for (int j = 0; j < 4; ++j) {
      const unsigned short* bp = &Bs[buf][wn * 64 + j * 16 + l15];
#pragma unroll
      for (int s = 0; s < 16; ++s)            // N = l15, K in table order
        b[j].h[s] = bp[((s >> 3) * 16 + h2 * 8 + (s & 7)) * BRS];
    }
#pragma unroll
    for (int i = 0; i < 4; ++i)
#pragma unroll
      for (int j = 0; j < 4; ++j)
        acc[i][j] = wmma_bf16(a[i].v, b[j].v, acc[i][j]);

    __syncthreads();                   // reads done before buf is restaged
  }

#pragma unroll
  for (int i = 0; i < 4; ++i)
#pragma unroll
    for (int j = 0; j < 4; ++j)
#pragma unroll
      for (int r = 0; r < 8; ++r) {
        int row = row0 + wm * 64 + i * 16 + h2 * 8 + r;  // C layout: VGPR r
        int col = col0 + wn * 64 + j * 16 + l15;
        float v = acc[i][j][r];
        if (MODE == 0)
          ((unsigned short*)outp)[(size_t)row * N + col] = f2bf(v);
        else if (MODE == 1)
          ((unsigned short*)outp)[(size_t)col * M + row] = f2bf(v);
        else
          ((float*)outp)[(size_t)row * N + col] = v;
      }
}

// ---------------- stage 2: flash sliding-window attention ----------------
// 1 wave per (head, 16-query tile). Qb:[S][DM] bf16 (pre-scaled),
// Kt:[DM][S] bf16, Vb:[S][DM] bf16 -> Ctx:[S][DM] bf16.
__global__ __launch_bounds__(32)
void attn_kernel(const unsigned short* __restrict__ Qb,
                 const unsigned short* __restrict__ Kt,
                 const unsigned short* __restrict__ Vb,
                 unsigned short* __restrict__ Ctx) {
  __shared__ alignas(32) unsigned short P[16][32];

  const int lane = threadIdx.x;
  const int h2   = lane >> 4;
  const int l15  = lane & 15;
  const int h    = blockIdx.y;
  const int qb   = blockIdx.x * 16;

  // Q fragments once: A 16x32 chunks over head dim (d = c*32 .. c*32+31)
  FragA aq[2];
#pragma unroll
  for (int c = 0; c < 2; ++c) {
    const unsigned short* qp =
        Qb + (size_t)(qb + l15) * D_MODEL + h * HEAD_DIM + c * 32 + h2 * 8;
    aq[c].p[0] = *(const U16x8*)qp;
    aq[c].p[1] = *(const U16x8*)(qp + 16);
  }

  v8f o[4];
  float m[8], lsum[8];
#pragma unroll
  for (int n = 0; n < 4; ++n) o[n] = {};
#pragma unroll
  for (int r = 0; r < 8; ++r) { m[r] = NEG_BIG; lsum[r] = 0.f; }

  const int jb0 = qb - HALF_W;
  const int it0 = jb0 < 0 ? ((-jb0) >> 5) : 0;               // first overlapping
  const int it1 = imin(16, (S_LEN - 1 - jb0) >> 5);          // last overlapping

  for (int it = it0; it <= it1; ++it) {
    const int jb = jb0 + 32 * it;                            // 32-key chunk
    v8f sf[2];
#pragma unroll
    for (int t = 0; t < 2; ++t) {
      const int jt  = jb + t * 16;
      const int jtc = imin(imax(jt, 0), S_LEN - 16);         // in-bounds addr
      const unsigned short* kp0 =
          Kt + (size_t)(h * HEAD_DIM) * S_LEN + jtc + l15;  // key = l15, d in table order
      const unsigned short* kp1 =
          Kt + (size_t)(h * HEAD_DIM + 32) * S_LEN + jtc + l15;
      FragA bk0, bk1;
#pragma unroll
      for (int e = 0; e < 16; ++e) {
        const int kd = (e >> 3) * 16 + h2 * 8 + (e & 7);
        bk0.h[e] = kp0[(size_t)kd * S_LEN];
        bk1.h[e] = kp1[(size_t)kd * S_LEN];
      }
      v8f s = {};
      s = wmma_bf16(aq[0].v, bk0.v, s);
      s = wmma_bf16(aq[1].v, bk1.v, s);
      sf[t] = s;
    }

    // mask (unclamped key index kills OOB/dup/out-of-window) + row max
    float rmax[8];
#pragma unroll
    for (int r = 0; r < 8; ++r) {
      const int q  = qb + h2 * 8 + r;
      const int j0 = jb + l15;
      const int j1 = jb + 16 + l15;
      bool v0 = (j0 >= 0) && (j0 < S_LEN) && (j0 >= q - HALF_W) && (j0 <= q + HALF_W);
      bool v1 = (j1 >= 0) && (j1 < S_LEN) && (j1 >= q - HALF_W) && (j1 <= q + HALF_W);
      float s0 = v0 ? sf[0][r] : NEG_BIG;
      float s1 = v1 ? sf[1][r] : NEG_BIG;
      sf[0][r] = s0; sf[1][r] = s1;
      float v = fmaxf(s0, s1);
      v = fmaxf(v, __shfl_xor(v, 1));
      v = fmaxf(v, __shfl_xor(v, 2));
      v = fmaxf(v, __shfl_xor(v, 4));
      v = fmaxf(v, __shfl_xor(v, 8));   // reduce over the 16-lane half = row
      rmax[r] = v;
    }

    float f[8];
#pragma unroll
    for (int r = 0; r < 8; ++r) {
      float nm = fmaxf(m[r], rmax[r]);
      f[r] = __expf(m[r] - nm);
      m[r] = nm;
      float p0 = __expf(sf[0][r] - nm);
      float p1 = __expf(sf[1][r] - nm);
      float rs = p0 + p1;
      rs += __shfl_xor(rs, 1);
      rs += __shfl_xor(rs, 2);
      rs += __shfl_xor(rs, 4);
      rs += __shfl_xor(rs, 8);
      lsum[r] = lsum[r] * f[r] + rs;
      P[h2 * 8 + r][l15]      = f2bf(p0);   // C-frag -> row-major P in LDS
      P[h2 * 8 + r][16 + l15] = f2bf(p1);
    }
#pragma unroll
    for (int n = 0; n < 4; ++n)
#pragma unroll
      for (int r = 0; r < 8; ++r) o[n][r] *= f[r];

    __syncthreads();  // single-wave block: just orders LDS write -> read

    FragA pa;         // re-read P as A fragment (16 q x 32 keys)
    const unsigned short* pp = &P[l15][h2 * 8];
    pa.p[0] = *(const U16x8*)pp;
    pa.p[1] = *(const U16x8*)(pp + 16);

#pragma unroll
    for (int n = 0; n < 4; ++n) {
      FragA vb;
#pragma unroll
      for (int e = 0; e < 16; ++e) {
        const int key = imin(imax(jb + (e >> 3) * 16 + h2 * 8 + (e & 7), 0), S_LEN - 1);
        vb.h[e] = Vb[(size_t)key * D_MODEL + h * HEAD_DIM + n * 16 + l15];
      }
      o[n] = wmma_bf16(pa.v, vb.v, o[n]);
    }
    __syncthreads();
  }

  // normalize + store context (bf16, natural [S][DM] layout)
#pragma unroll
  for (int r = 0; r < 8; ++r) {
    float inv = lsum[r] > 0.f ? 1.f / lsum[r] : 0.f;
    const int row = qb + h2 * 8 + r;
#pragma unroll
    for (int n = 0; n < 4; ++n)
      Ctx[(size_t)row * D_MODEL + h * HEAD_DIM + n * 16 + l15] =
          f2bf(o[n][r] * inv);
  }
}

// ---------------- host launcher ----------------
extern "C" void kernel_launch(void* const* d_in, const int* in_sizes, int n_in,
                              void* d_out, int out_size, void* d_ws, size_t ws_size,
                              hipStream_t stream) {
  (void)in_sizes; (void)n_in; (void)out_size; (void)ws_size;
  const float* x  = (const float*)d_in[0];
  const float* wq = (const float*)d_in[1];
  const float* wk = (const float*)d_in[2];
  const float* wv = (const float*)d_in[3];
  const float* wo = (const float*)d_in[4];

  const size_t SD = (size_t)S_LEN * D_MODEL;     // 4M elems
  const size_t DD = (size_t)D_MODEL * D_MODEL;   // 1M elems
  unsigned short* Xb  = (unsigned short*)d_ws;   // bf16 x            [S][DM]
  unsigned short* Wtq = Xb  + SD;                // (wq^T)*scale      [DM][DM]
  unsigned short* Wtk = Wtq + DD;
  unsigned short* Wtv = Wtk + DD;
  unsigned short* Wto = Wtv + DD;
  unsigned short* Qb  = Wto + DD;                // Q (pre-scaled)    [S][DM]
  unsigned short* Kt  = Qb  + SD;                // K transposed      [DM][S]
  unsigned short* Vb  = Kt  + SD;                // V                 [S][DM]
  unsigned short* Ctx = Vb  + SD;                // attention output  [S][DM]

  cast_f32_bf16<<<dim3((unsigned)(SD / 256)), 256, 0, stream>>>(x, Xb, (int)SD);
  cast_transpose_bf16<<<dim3((unsigned)(DD / 256)), 256, 0, stream>>>(wq, Wtq, ATT_SCALE);
  cast_transpose_bf16<<<dim3((unsigned)(DD / 256)), 256, 0, stream>>>(wk, Wtk, 1.0f);
  cast_transpose_bf16<<<dim3((unsigned)(DD / 256)), 256, 0, stream>>>(wv, Wtv, 1.0f);
  cast_transpose_bf16<<<dim3((unsigned)(DD / 256)), 256, 0, stream>>>(wo, Wto, 1.0f);

  dim3 gg(S_LEN / 128, D_MODEL / 128);
  gemm_bf16<0><<<gg, 128, 0, stream>>>(Xb, Wtq, Qb, S_LEN, D_MODEL, D_MODEL);
  gemm_bf16<1><<<gg, 128, 0, stream>>>(Xb, Wtk, Kt, S_LEN, D_MODEL, D_MODEL);
  gemm_bf16<0><<<gg, 128, 0, stream>>>(Xb, Wtv, Vb, S_LEN, D_MODEL, D_MODEL);

  attn_kernel<<<dim3(S_LEN / 16, N_HEADS), 32, 0, stream>>>(Qb, Kt, Vb, Ctx);

  gemm_bf16<2><<<gg, 128, 0, stream>>>(Ctx, Wto, d_out, S_LEN, D_MODEL, D_MODEL);
}
